// hyperCL_87299505259187
// MI455X (gfx1250) — compile-verified, matrix operand order checked by reading, not run on hardware
//
#include <hip/hip_runtime.h>

typedef __attribute__((ext_vector_type(16))) _Float16 v16h;
typedef __attribute__((ext_vector_type(8)))  float    v8f;

// Guaranteed-native no-return f32 VMEM atomic (tracked by STOREcnt).
__device__ __forceinline__ void atomAddF(float* p, float v) {
  asm volatile("global_atomic_add_f32 %0, %1, off"
               :
               : "v"(p), "v"(v)
               : "memory");
}

__device__ __forceinline__ float preluf(float v, float a) {
  return v >= 0.0f ? v : a * v;
}

// ---------------------------------------------------------------------------
// GEMM: Y[R,64] = X[R,64] @ W[64,64]  (fp16 WMMA, fp32 accumulate)
// One wave = one 16-row tile across all 64 output columns.
// All 8 B-tiles preloaded -> 8 back-to-back v_wmma issues per tile.
// ---------------------------------------------------------------------------
__global__ void __launch_bounds__(256) gemm64_wmma(
    const float* __restrict__ X, const float* __restrict__ W,
    float* __restrict__ Y, int rowTiles) {
  __shared__ _Float16 Wt[64 * 72];  // W transposed, padded pitch 72 halves
  int tid = threadIdx.x;
  for (int i = tid; i < 64 * 64; i += 256) {
    int k = i >> 6, n = i & 63;
    Wt[n * 72 + k] = (_Float16)W[i];
  }
  __syncthreads();

  int wave = tid >> 5, lane = tid & 31;
  int tile = blockIdx.x * 8 + wave;
  if (tile >= rowTiles) return;          // wave-uniform: EXEC stays all-1s

  int row0 = tile * 16;
  int m    = lane & 15;
  int kb   = (lane < 16) ? 0 : 8;        // CDNA5 16-bit A layout K striping
  int row  = row0 + m;

  // A tiles: kt=0 -> K 0..31, kt=1 -> K 32..63
  v16h a0, a1;
  const float* xr = X + (size_t)row * 64;
  #pragma unroll
  for (int kt = 0; kt < 2; ++kt) {
    const float* p = xr + kt * 32 + kb;
    v16h a;
    #pragma unroll
    for (int j = 0; j < 8; ++j) {
      a[j]     = (_Float16)p[j];         // K = kb + j
      a[8 + j] = (_Float16)p[16 + j];    // K = kb + 16 + j
    }
    if (kt == 0) a0 = a; else a1 = a;
  }

  // Preload all B tiles (4 N-tiles x 2 K-tiles)
  v16h bt0[4], bt1[4];
  #pragma unroll
  for (int nt = 0; nt < 4; ++nt) {
    const _Float16* w0 = Wt + (nt * 16 + m) * 72 + kb;  // column nt*16+m of W
    #pragma unroll
    for (int j = 0; j < 8; ++j) {
      bt0[nt][j]     = w0[j];       bt0[nt][8 + j] = w0[16 + j];  // K 0..31
      bt1[nt][j]     = w0[32 + j];  bt1[nt][8 + j] = w0[48 + j];  // K 32..63
    }
  }

  v8f c[4];
  #pragma unroll
  for (int nt = 0; nt < 4; ++nt) c[nt] = (v8f){};
  #pragma unroll
  for (int nt = 0; nt < 4; ++nt)
    c[nt] = __builtin_amdgcn_wmma_f32_16x16x32_f16(false, a0, false, bt0[nt],
                                                   (short)0, c[nt], false, false);
  #pragma unroll
  for (int nt = 0; nt < 4; ++nt)
    c[nt] = __builtin_amdgcn_wmma_f32_16x16x32_f16(false, a1, false, bt1[nt],
                                                   (short)0, c[nt], false, false);

  // C/D layout: VGPR v -> row (v or 8+v), N = lane&15
  int mBase = row0 + ((lane < 16) ? 0 : 8);
  #pragma unroll
  for (int nt = 0; nt < 4; ++nt) {
    float* y = Y + (size_t)mBase * 64 + nt * 16 + m;
    #pragma unroll
    for (int v = 0; v < 8; ++v) y[(size_t)v * 64] = c[nt][v];
  }
}

// ---------------------------------------------------------------------------
// Degrees (index-only, shared by both layers). Self-loops folded analytically:
// Dn starts at 1 (every node has one self-loop), De[e>=NE] == 1 exactly.
// ---------------------------------------------------------------------------
__global__ void deg_init(float* Dn, float* De, int NN, int NE, int ET) {
  int i = blockIdx.x * blockDim.x + threadIdx.x;
  if (i < NN) Dn[i] = 1.0f;
  if (i < ET) De[i] = (i < NE) ? 0.0f : 1.0f;
}

__global__ void deg_accum(const int* __restrict__ ni, const int* __restrict__ ei,
                          float* Dn, float* De, int nnz) {
  int i = blockIdx.x * blockDim.x + threadIdx.x;
  if (i < nnz) {
    atomAddF(&Dn[ni[i]], 1.0f);
    atomAddF(&De[ei[i]], 1.0f);
  }
}

__global__ void deg_inv(float* Dn, float* De, int NN, int ET) {
  int i = blockIdx.x * blockDim.x + threadIdx.x;
  if (i < NN) Dn[i] = 1.0f / Dn[i];                       // Dn >= 1 always
  if (i < ET) { float v = De[i]; De[i] = (v > 0.0f) ? 1.0f / v : 0.0f; }
}

// ---------------------------------------------------------------------------
// e_raw init: zero real edges, copy self-loop row h[i] into edge NE+i (unique)
// ---------------------------------------------------------------------------
__global__ void e_init(float4* __restrict__ e_raw, const float4* __restrict__ h,
                       int NE, int ET) {
  int i = blockIdx.x * blockDim.x + threadIdx.x;
  if (i >= ET * 16) return;
  int e = i >> 4;
  float4 z = make_float4(0.f, 0.f, 0.f, 0.f);
  e_raw[i] = (e < NE) ? z : h[i - NE * 16];
}

// n_raw init: self-loop contribution g[NE+i] -> node i (unique per row)
__global__ void n_init(float4* __restrict__ n_raw, const float4* __restrict__ g,
                       int NE, int NN) {
  int i = blockIdx.x * blockDim.x + threadIdx.x;
  if (i >= NN * 16) return;
  n_raw[i] = g[(size_t)NE * 16 + i];
}

// ---------------------------------------------------------------------------
// Scatter-add: one wave per nnz entry, 2 floats/lane -> 64 f32 atomics/entry.
// dst[didx[j]][:] += src[gidx[j]][:]
// Indices are wave-uniform -> readfirstlane moves row bases into SGPRs.
// ---------------------------------------------------------------------------
__global__ void scatter_add64(const int* __restrict__ gidx, const int* __restrict__ didx,
                              const float* __restrict__ src, float* __restrict__ dst,
                              int nnz) {
  long long t = (long long)blockIdx.x * blockDim.x + threadIdx.x;
  int j = (int)(t >> 5);                 // wave-uniform entry index
  if (j >= nnz) return;
  int lane = (int)t & 31;
  int gi = __builtin_amdgcn_readfirstlane(gidx[j]);
  int di = __builtin_amdgcn_readfirstlane(didx[j]);
  const float* s = src + (size_t)gi * 64 + lane * 2;
  float*       d = dst + (size_t)di * 64 + lane * 2;
  float2 v = *(const float2*)s;
  atomAddF(d,     v.x);
  atomAddF(d + 1, v.y);
}

// Fused epilogues: factored norm * sum + bias, then PReLU
__global__ void e_fin(float4* e, const float* __restrict__ De_inv,
                      const float4* __restrict__ b, const float* __restrict__ alphaP,
                      int ET) {
  int i = blockIdx.x * blockDim.x + threadIdx.x;
  if (i >= ET * 16) return;
  float a = alphaP[0];
  float s = De_inv[i >> 4];
  float4 bb = b[i & 15];
  float4 v = e[i];
  v.x = preluf(s * v.x + bb.x, a);
  v.y = preluf(s * v.y + bb.y, a);
  v.z = preluf(s * v.z + bb.z, a);
  v.w = preluf(s * v.w + bb.w, a);
  e[i] = v;
}

__global__ void n_fin(float4* out, const float4* n_raw,   // may alias (in-place)
                      const float* __restrict__ Dn_inv, const float4* __restrict__ b,
                      const float* __restrict__ alphaP, int NN) {
  int i = blockIdx.x * blockDim.x + threadIdx.x;
  if (i >= NN * 16) return;
  float a = alphaP[0];
  float s = Dn_inv[i >> 4];
  float4 bb = b[i & 15];
  float4 v = n_raw[i];
  float4 o;
  o.x = preluf(s * v.x + bb.x, a);   // conv bias-add then encoder-level PReLU
  o.y = preluf(s * v.y + bb.y, a);
  o.z = preluf(s * v.z + bb.z, a);
  o.w = preluf(s * v.w + bb.w, a);
  out[i] = o;
}

static inline int blocksFor(long long n, int bs) { return (int)((n + bs - 1) / bs); }

extern "C" void kernel_launch(void* const* d_in, const int* in_sizes, int n_in,
                              void* d_out, int out_size, void* d_ws, size_t ws_size,
                              hipStream_t stream) {
  const float* x      = (const float*)d_in[0];
  const int*   hei    = (const int*)d_in[1];
  const float* W1n2e  = (const float*)d_in[2];
  const float* b1n2e  = (const float*)d_in[3];
  const float* W1e2n  = (const float*)d_in[4];
  const float* b1e2n  = (const float*)d_in[5];
  const float* W2n2e  = (const float*)d_in[6];
  const float* b2n2e  = (const float*)d_in[7];
  const float* W2e2n  = (const float*)d_in[8];
  const float* b2e2n  = (const float*)d_in[9];
  const float* alpha  = (const float*)d_in[10];

  const int NN  = in_sizes[0] / 64;   // 100000
  const int nnz = in_sizes[1] / 2;    // 1600000
  const int NE  = 20000;              // N_HEDGES: reference constant (device-scalar only)
  const int ET  = NE + NN;            // 120000

  const int* ni = hei;                // hyperedge_index row 0
  const int* ei = hei + nnz;          // hyperedge_index row 1

  // workspace layout (floats)
  float* bufA = (float*)d_ws;                 // h / g         [ET*64]
  float* bufB = bufA + (size_t)ET * 64;       // e_raw / e     [ET*64]
  float* bufC = bufB + (size_t)ET * 64;       // n_raw / h1out [NN*64]
  float* Dn   = bufC + (size_t)NN * 64;       // [NN]  -> Dn_inv in place
  float* De   = Dn + NN;                      // [ET]  -> De_inv in place

  float* out = (float*)d_out;

  const int BS = 256;
  const int tilesN = NN / 16;   // 6250
  const int tilesE = ET / 16;   // 7500
  const int gemmBlkN = blocksFor(tilesN, 8);
  const int gemmBlkE = blocksFor(tilesE, 8);
  const long long scatThreads = (long long)nnz * 32;

  // ---- degrees (once) ----
  deg_init <<<blocksFor(ET, BS), BS, 0, stream>>>(Dn, De, NN, NE, ET);
  deg_accum<<<blocksFor(nnz, BS), BS, 0, stream>>>(ni, ei, Dn, De, nnz);
  deg_inv  <<<blocksFor(ET, BS), BS, 0, stream>>>(Dn, De, NN, ET);

  // ==================== layer 1 ====================
  gemm64_wmma<<<gemmBlkN, BS, 0, stream>>>(x, W1n2e, bufA, tilesN);                 // h = x @ W1_n2e
  e_init <<<blocksFor((long long)ET * 16, BS), BS, 0, stream>>>((float4*)bufB, (const float4*)bufA, NE, ET);
  scatter_add64<<<blocksFor(scatThreads, BS), BS, 0, stream>>>(ni, ei, bufA, bufB, nnz);
  e_fin  <<<blocksFor((long long)ET * 16, BS), BS, 0, stream>>>((float4*)bufB, De, (const float4*)b1n2e, alpha, ET);
  gemm64_wmma<<<gemmBlkE, BS, 0, stream>>>(bufB, W1e2n, bufA, tilesE);              // g = e @ W1_e2n
  n_init <<<blocksFor((long long)NN * 16, BS), BS, 0, stream>>>((float4*)bufC, (const float4*)bufA, NE, NN);
  scatter_add64<<<blocksFor(scatThreads, BS), BS, 0, stream>>>(ei, ni, bufA, bufC, nnz);
  n_fin  <<<blocksFor((long long)NN * 16, BS), BS, 0, stream>>>((float4*)bufC, (const float4*)bufC, Dn, (const float4*)b1e2n, alpha, NN);

  // ==================== layer 2 ====================
  gemm64_wmma<<<gemmBlkN, BS, 0, stream>>>(bufC, W2n2e, bufA, tilesN);              // h2 = h1 @ W2_n2e
  e_init <<<blocksFor((long long)ET * 16, BS), BS, 0, stream>>>((float4*)bufB, (const float4*)bufA, NE, ET);
  scatter_add64<<<blocksFor(scatThreads, BS), BS, 0, stream>>>(ni, ei, bufA, bufB, nnz);
  e_fin  <<<blocksFor((long long)ET * 16, BS), BS, 0, stream>>>((float4*)bufB, De, (const float4*)b2n2e, alpha, ET);
  gemm64_wmma<<<gemmBlkE, BS, 0, stream>>>(bufB, W2e2n, bufA, tilesE);              // g2 = e2 @ W2_e2n
  n_init <<<blocksFor((long long)NN * 16, BS), BS, 0, stream>>>((float4*)bufC, (const float4*)bufA, NE, NN);
  scatter_add64<<<blocksFor(scatThreads, BS), BS, 0, stream>>>(ei, ni, bufA, bufC, nnz);
  n_fin  <<<blocksFor((long long)NN * 16, BS), BS, 0, stream>>>((float4*)out, (const float4*)bufC, Dn, (const float4*)b2e2n, alpha, NN);
}
